// AttentionConvFull_1082331759189
// MI455X (gfx1250) — matrix operand order repaired and checked by reading, not compiled, not run on hardware
//
#include <hip/hip_runtime.h>
#include <hip/hip_bf16.h>

typedef __attribute__((ext_vector_type(16))) _Float16 v16h;
typedef __attribute__((ext_vector_type(8)))  float    v8f;
typedef __attribute__((ext_vector_type(4)))  float    v4f;

#define CG     32      // channels per group
#define TILE   8       // output tile edge
#define HALO   12      // TILE + 2*pad
#define NHALO  144     // HALO*HALO
#define XSTR   34      // f16 LDS row stride for x (32 + pad, even for pair loads)
#define KSTR   33      // f32 LDS row stride for k/v (bank-conflict pad)

// Fused: grouped 1x1 q/k/v projections (WMMA f16->f32) + per-channel 5x5
// windowed softmax attention. One pass over x, one pass writing out.
__global__ __launch_bounds__(256)
void attnconv_fused_wmma_kernel(const float* __restrict__ x,
                                const float* __restrict__ wq,
                                const float* __restrict__ wk,
                                const float* __restrict__ wv,
                                const float* __restrict__ rel,
                                const float* __restrict__ qemb,
                                float* __restrict__ out) {
  __shared__ _Float16 xls[NHALO * XSTR];   // x halo tile, f16
  __shared__ float    kls[NHALO * KSTR];   // k projection over halo
  __shared__ float    vls[NHALO * KSTR];   // v projection over halo
  __shared__ float    rls[CG * 25];        // rel_emb for this group

  const int tid = threadIdx.x;
  const int ty  = blockIdx.x / 7, tx = blockIdx.x % 7;
  const int g   = blockIdx.y;
  const int b   = blockIdx.z;
  const int y0  = ty * TILE, x0 = tx * TILE;

  // ---------- Phase A: stage x halo (zero-padded) as f16 + rel_emb ----------
  for (int q4 = tid; q4 < NHALO * (CG / 4); q4 += 256) {
    int pos = q4 >> 3;
    int c4  = (q4 & 7) << 2;
    int hy  = pos / HALO, hx = pos % HALO;
    int gy  = y0 - 2 + hy, gx = x0 - 2 + hx;
    v4f v = {0.f, 0.f, 0.f, 0.f};
    if (gy >= 0 && gy < 56 && gx >= 0 && gx < 56)
      v = *(const v4f*)(x + (((size_t)(b * 56 + gy) * 56 + gx) * 256 + g * CG + c4));
    _Float16* dst = &xls[pos * XSTR + c4];
    dst[0] = (_Float16)v.x; dst[1] = (_Float16)v.y;
    dst[2] = (_Float16)v.z; dst[3] = (_Float16)v.w;
  }
  for (int i = tid; i < CG * 25; i += 256)
    rls[i] = rel[(size_t)g * (CG * 25) + i];
  __syncthreads();

  const int wave    = tid >> 5;
  const int lane    = tid & 31;
  const int n       = lane & 15;   // B/D column (N); also A row (M)
  const int hi      = lane >> 4;   // which K-slice this lane holds
  const int colhalf = wave >> 2;   // 16-channel half of Cg=32
  const int rowtile = wave & 3;    // 16-position row tile of the 64 outputs

  // ISA layouts (CDNA5 ISA): 16-bit A 16x32 stripes each 16-K block as
  // lo8/hi8 across lane halves per 4-VGPR group; 16-bit B uses the same
  // K order as A (lo8/hi8 per 16-K block across lane halves).
  auto kbA = [&](int r) { return 2 * r + ((r & 4) ? 8 : 0) + 8 * hi; };
  auto kbB = [&](int r) { return 2 * r + ((r & 4) ? 8 : 0) + 8 * hi; };

  // A fragment: 16 halo positions x 32 input channels from LDS (f16).
  auto loadA = [&](int pos) {
    const _Float16* xr = &xls[pos * XSTR];
    v16h a;
#pragma unroll
    for (int r = 0; r < 8; ++r) {
      int kb = kbA(r);
      a[2 * r]     = xr[kb];
      a[2 * r + 1] = xr[kb + 1];
    }
    return a;
  };
  // B fragment: 32 input channels x 16 output channels; B[k][o] = w[g][o][k].
  auto loadB = [&](const float* wmat, int ch) {
    const float* wr = wmat + ((size_t)g * CG + ch * 16 + n) * CG;
    v16h bv;
#pragma unroll
    for (int r = 0; r < 8; ++r) {
      int kb = kbB(r);
      bv[2 * r]     = (_Float16)wr[kb];
      bv[2 * r + 1] = (_Float16)wr[kb + 1];
    }
    return bv;
  };

  // ---------- Phase B: projections via v_wmma_f32_16x16x32_f16 ----------
  // q for this wave's 16-pos x 16-ch tile, kept in registers.
  v8f accq = {};
  {
    int pc = rowtile * 16 + n;                         // center pos 0..63
    int hq = ((pc >> 3) + 2) * HALO + (pc & 7) + 2;    // halo index of center
    v16h aq = loadA(hq);
    v16h bq = loadB(wq, colhalf);
    accq = __builtin_amdgcn_wmma_f32_16x16x32_f16(
        false, aq, false, bq, (short)0, accq, false, false);
  }

  // k and v over all 144 halo positions: 9 row tiles x 2 col halves x {k,v}
  for (int j = wave; j < 36; j += 8) {      // uniform per wave -> EXEC all 1s
    int isv = (j >= 18);
    int jj  = isv ? j - 18 : j;
    int rt  = jj >> 1, ch = jj & 1;
    v16h a  = loadA(rt * 16 + n);
    v16h bw = loadB(isv ? wv : wk, ch);
    v8f d = {};
    d = __builtin_amdgcn_wmma_f32_16x16x32_f16(
        false, a, false, bw, (short)0, d, false, false);
    float* dst = isv ? vls : kls;
#pragma unroll
    for (int r = 0; r < 8; ++r)
      dst[(rt * 16 + r + 8 * hi) * KSTR + ch * 16 + n] = d[r];
  }
  __syncthreads();

  // ---------- Phase C: per-channel 5x5 softmax attention ----------
  const int   cg = colhalf * 16 + n;
  const float qe = qemb[g * CG + cg];
#pragma unroll
  for (int r = 0; r < 8; ++r) {
    float qv = accq[r] + qe;
    int p  = rowtile * 16 + r + 8 * hi;   // this lane's center position
    int py = p >> 3, px = p & 7;
    float lg[25];
    float mx = -3.0e38f;
#pragma unroll
    for (int i = 0; i < 25; ++i) {
      int hp = (py + i / 5) * HALO + (px + i % 5);
      float kv = kls[hp * KSTR + cg] + rls[cg * 25 + i];
      float l  = qv * kv;
      lg[i] = l;
      mx = fmaxf(mx, l);
    }
    float s = 0.f, o = 0.f;
#pragma unroll
    for (int i = 0; i < 25; ++i) {
      int hp = (py + i / 5) * HALO + (px + i % 5);
      float e = __expf(lg[i] - mx);
      s += e;
      o += e * vls[hp * KSTR + cg];
    }
    int gy = y0 + py, gx = x0 + px;
    out[(((size_t)(b * 56 + gy) * 56 + gx) * 256) + g * CG + cg] = o / s;
  }
}

extern "C" void kernel_launch(void* const* d_in, const int* in_sizes, int n_in,
                              void* d_out, int out_size, void* d_ws, size_t ws_size,
                              hipStream_t stream) {
  const float* x    = (const float*)d_in[0];
  const float* wq   = (const float*)d_in[1];
  const float* wk   = (const float*)d_in[2];
  const float* wv   = (const float*)d_in[3];
  const float* rel  = (const float*)d_in[4];
  const float* qemb = (const float*)d_in[5];
  float* out = (float*)d_out;

  dim3 grid(7 * 7, 8, 4);   // (spatial tiles, groups, batch)
  dim3 block(256);
  attnconv_fused_wmma_kernel<<<grid, block, 0, stream>>>(x, wq, wk, wv, rel,
                                                         qemb, out);
}
